// D_GCN_79766132621679
// MI455X (gfx1250) — matrix operand order repaired and checked by reading, not compiled, not run on hardware
//
#include <hip/hip_runtime.h>
#include <hip/hip_bf16.h>
#include <stddef.h>

// ---------------------------------------------------------------------------
// D_GCN on MI455X (gfx1250): Chebyshev diffusion GCN.
//   B=2, T=8, N=4096, F=32, O=32, K=3
// Dominant cost: two (4096x4096)x(4096x32) GEMMs per (b,t) pair -> 34.4 GFLOP.
// Strategy: bf16 WMMA (v_wmma_f32_16x16x32_bf16) with fp32 accumulation.
//  - A converted to bf16 once; streamed from HBM (134 MB/pass with the 2-way
//    column split ~ 5.7 us @ 23.3 TB/s, below the ~14 us bf16 compute time).
//  - B panel staged per-workgroup into LDS in 8 KB k-slices with
//    GLOBAL_LOAD_ASYNC_TO_LDS_B128 (ASYNCcnt), TRIPLE-buffered: slice k+2 is
//    issued one full iteration ahead; s_wait_asynccnt<=2 exploits in-order
//    async completion.
//  - k-loop unrolled by 2 with ping-pong A-fragment registers (no rotation
//    movs, no WMMA source-overwrite hazard nops); B-fragments rotate through
//    two register buffers so waitcnts stay relaxed (dscnt/loadcnt <= 2).
//  - 2-way column split: 128 WGs / 1024 waves to cover the WGP array.
// ---------------------------------------------------------------------------

#define BB 2
#define TT 8
#define NN 4096
#define FF 32
#define OO 32
#define KDIM NN           // GEMM reduction dim
#define CC (TT * FF)      // 256 columns per batch (all t,f folded together)
#define KSTEPS (KDIM / 32)
#define COLS_PER_WG 128   // 2-way column split
#define NT (COLS_PER_WG / 16)       // 8 col-tiles per wave
#define SLICE_BYTES (COLS_PER_WG * 32 * 2)   // 8 KB per k-slice

typedef __attribute__((ext_vector_type(16))) __bf16 bf16x16;
typedef __attribute__((ext_vector_type(4)))  __bf16 bf16x4;
typedef __attribute__((ext_vector_type(8)))  float  floatx8;

struct alignas(16) U4 { unsigned int x, y, z, w; };
union ABfrag { U4 u[2]; bf16x16 v; };
struct alignas(16) F4 { float x, y, z, w; };

// ---------------------------------------------------------------------------
// A_q fp32 -> bf16, same row-major [b][n][k] layout. 4 elems / thread.
// ---------------------------------------------------------------------------
__global__ void __launch_bounds__(256) convertA_kernel(
    const float* __restrict__ src, __bf16* __restrict__ dst, int n4) {
  int i = blockIdx.x * 256 + threadIdx.x;
  if (i >= n4) return;
  F4 v = *(const F4*)(src + (size_t)i * 4);
  bf16x4 o = { (__bf16)v.x, (__bf16)v.y, (__bf16)v.z, (__bf16)v.w };
  *(bf16x4*)(dst + (size_t)i * 4) = o;
}

// ---------------------------------------------------------------------------
// X (b,t,n,f) fp32 -> bf16 column-major [b][c = t*32+f][k = n].
// Column-major B means the LDS k-slices are 64B-contiguous per column.
// ---------------------------------------------------------------------------
__global__ void __launch_bounds__(256) convertX_kernel(
    const float* __restrict__ src, __bf16* __restrict__ dst) {
  int i = blockIdx.x * 256 + threadIdx.x;      // 2^21 elements, exact grid
  int f = i & 31;
  int n = (i >> 5) & (NN - 1);
  int t = (i >> 17) & (TT - 1);
  int b = i >> 20;
  dst[((size_t)b * CC + t * FF + f) * KDIM + n] = (__bf16)src[i];
}

// ---------------------------------------------------------------------------
// Async copy of one thread's 32B share of an 8 KB k-slice (2x b128).
// Thread tid owns column c0 + (tid>>1), half (tid&1) of its 64B run.
// ---------------------------------------------------------------------------
__device__ __forceinline__ void async_copy_slice(unsigned ldsDst,
                                                 unsigned long long gSrc) {
  asm volatile("global_load_async_to_lds_b128 %0, %1, off"
               :: "v"(ldsDst), "v"(gSrc) : "memory");
  asm volatile("global_load_async_to_lds_b128 %0, %1, off"
               :: "v"(ldsDst + 16u), "v"(gSrc + 16u) : "memory");
}

// One k-step: wait for slice KS, issue slice KS+2, prefetch next A fragment
// into AF_LOAD while running 8 WMMAs off AF_USE against LDS B-fragments.
#define KSTEP_BODY(KS, AF_USE, AF_LOAD)                                       \
  do {                                                                        \
    const int k0_ = (KS) * 32;                                                \
    if ((KS) + 1 < KSTEPS)                                                    \
      asm volatile("s_wait_asynccnt 0x2" ::: "memory");                       \
    else                                                                      \
      asm volatile("s_wait_asynccnt 0x0" ::: "memory");                       \
    __syncthreads();                                                          \
    if ((KS) + 2 < KSTEPS) {                                                  \
      int nb_ = bufIdx + 2; if (nb_ >= 3) nb_ -= 3;                           \
      async_copy_slice(ldsBase + (unsigned)nb_ * SLICE_BYTES,                 \
                       (unsigned long long)(gB + (size_t)((KS) + 2) * 64));   \
    }                                                                         \
    /* A fragment for k-step KS+1; last step reads <=64B past the row end */  \
    /* which is still inside d_ws. */                                         \
    AF_LOAD.u[0] = *(const U4*)(ap + k0_ + 32);                               \
    AF_LOAD.u[1] = *(const U4*)(ap + k0_ + 48);                               \
    __builtin_prefetch(ap + k0_ + 256, 0, 1);                                 \
    const char* p0_ = &sB[bufIdx][0] + lane16 * 64 + half * 16;               \
    ABfrag bfr_[2];                                                           \
    bfr_[0].u[0] = *(const U4*)(p0_);                                         \
    bfr_[0].u[1] = *(const U4*)(p0_ + 32);                                    \
    _Pragma("unroll")                                                         \
    for (int ct = 0; ct < NT; ++ct) {                                         \
      const char* pn_ = p0_ + (((ct + 1) & (NT - 1)) * 16 * 64);              \
      bfr_[(ct + 1) & 1].u[0] = *(const U4*)pn_;        /* ds_load_b128 */    \
      bfr_[(ct + 1) & 1].u[1] = *(const U4*)(pn_ + 32); /* ds_load_b128 */    \
      acc[ct] = __builtin_amdgcn_wmma_f32_16x16x32_bf16(                      \
          false, AF_USE.v, false, bfr_[ct & 1].v,                             \
          (short)0, acc[ct], false, false);                                   \
    }                                                                         \
    ++bufIdx; if (bufIdx >= 3) bufIdx = 0;                                    \
  } while (0)

// ---------------------------------------------------------------------------
// C[b] = A[b] @ B[b] : A row-major bf16 [b][N][K], B col-major bf16 [b][CC][K].
// Wave = 16 output rows x 128 columns (8 col-tiles), K in steps of 32 via
// v_wmma_f32_16x16x32_bf16; 8 accumulators (64 VGPRs) + rotating operand
// buffers. B slices come from triple-buffered LDS fed by async loads.
// Register layouts per CDNA5 ISA 7.12.2:
//   A 16x32 bf16: lane=row(M), elems 0..7 = K(+0..7|+8..15), 8..15 = K(+16..)
//   B 32x16 bf16: lane=col(N), elems 0..7 = K(+0..7|+8..15), 8..15 = K(+16..), same as A
//   C 16x16 f32 : elem r -> M = r + 8*(lane/16), N = lane&15
// ---------------------------------------------------------------------------
__global__ void __launch_bounds__(256) gemm_diffuse_kernel(
    const __bf16* __restrict__ Abf,   // [b][N][K]
    const __bf16* __restrict__ Bt,    // [b][CC][K]
    __bf16* __restrict__ outBf,       // [b][CC][N] or nullptr
    float*  __restrict__ outF) {      // [b][CC][N] or nullptr
  __shared__ __align__(1024) char sB[3][SLICE_BYTES];

  const int tid    = threadIdx.x;
  const int colHalf= blockIdx.y;          // 2-way column split
  const int b      = blockIdx.z;
  const int wave   = tid >> 5;
  const int lid    = tid & 31;
  const int half   = lid >> 4;            // lane half selects K sub-range
  const int lane16 = lid & 15;
  const int m0     = (blockIdx.x * 8 + wave) * 16;
  const int c0     = colHalf * COLS_PER_WG;

  const __bf16* ap = Abf + ((size_t)b * NN + (m0 + lane16)) * KDIM + half * 8;

  // Per-thread async-copy bases: column c0 + (tid>>1), 32B half (tid&1).
  const char* gB = (const char*)(Bt + ((size_t)b * CC + c0 + (tid >> 1)) * KDIM)
                   + (tid & 1) * 32;
  unsigned ldsBase = (unsigned)(unsigned long long)(void*)&sB[0][0]
                   + (unsigned)(tid >> 1) * 64u + (unsigned)(tid & 1) * 32u;

  // Prologue: slices 0 and 1 in flight (2 async ops each, in-order done).
  async_copy_slice(ldsBase,               (unsigned long long)gB);
  async_copy_slice(ldsBase + SLICE_BYTES, (unsigned long long)(gB + 64));

  // First A fragment.
  ABfrag afA, afB;
  afA.u[0] = *(const U4*)(ap);
  afA.u[1] = *(const U4*)(ap + 16);

  floatx8 acc[NT];
#pragma unroll
  for (int ct = 0; ct < NT; ++ct) acc[ct] = (floatx8)0.0f;

  int bufIdx = 0;                         // sB buffer holding slice ks
  for (int ks = 0; ks < KSTEPS; ks += 2) {
    KSTEP_BODY(ks,     afA, afB);         // consume afA, load afB
    KSTEP_BODY(ks + 1, afB, afA);         // consume afB, load afA
  }

  // Epilogue: write column-major [b][c][n] (bf16 for next pass, f32 for Y).
#pragma unroll
  for (int ct = 0; ct < NT; ++ct) {
    int col  = c0 + ct * 16 + lane16;
    int row0 = m0 + half * 8;
    size_t base = ((size_t)b * CC + col) * NN + row0;
#pragma unroll
    for (int r = 0; r < 8; ++r) {
      float v = acc[ct][r];
      if (outBf) outBf[base + r] = (__bf16)v;
      if (outF)  outF[base + r]  = v;
    }
  }
}

// ---------------------------------------------------------------------------
// out[b,t,n,o] = relu( x0@Th0 + x1@Th1 + (2y - x0)@Th2 + bias ),
// Theta row index = f*3 + k (stack axis=-1 interleaving). Theta in LDS.
// ---------------------------------------------------------------------------
__global__ void __launch_bounds__(256) combine_kernel(
    const float*  __restrict__ X,     // (b,t,n,f) fp32
    const __bf16* __restrict__ X1t,   // [b][c][n] bf16
    const float*  __restrict__ Yt,    // [b][c][n] fp32 (A @ x1)
    const float*  __restrict__ Theta, // (96,32)
    const float*  __restrict__ bias,  // (32,)
    float* __restrict__ out) {        // (b,t,n,o)
  __shared__ float sTh[96 * OO + OO];
  for (int i = threadIdx.x; i < 96 * OO + OO; i += 256)
    sTh[i] = (i < 96 * OO) ? Theta[i] : bias[i - 96 * OO];
  __syncthreads();

  int tid = blockIdx.x * 256 + threadIdx.x;   // B*T*N = 65536, exact grid
  int b = tid >> 15;
  int t = (tid >> 12) & (TT - 1);
  int n = tid & (NN - 1);

  const float* xp  = X + (size_t)tid * FF;
  size_t cbase = ((size_t)b * CC + t * FF) * NN + n;

  float acc[OO];
#pragma unroll
  for (int o = 0; o < OO; ++o) acc[o] = sTh[96 * OO + o];

  for (int f = 0; f < FF; ++f) {
    float a0 = xp[f];
    float a1 = (float)X1t[cbase + (size_t)f * NN];
    float a2 = 2.0f * Yt[cbase + (size_t)f * NN] - a0;
    const float* t0 = sTh + (3 * f) * OO;
#pragma unroll
    for (int o = 0; o < OO; ++o)
      acc[o] = fmaf(a0, t0[o], fmaf(a1, t0[OO + o], fmaf(a2, t0[2 * OO + o], acc[o])));
  }

  float* op = out + (size_t)tid * OO;
#pragma unroll
  for (int o = 0; o < OO; ++o) op[o] = fmaxf(acc[o], 0.0f);
}

// ---------------------------------------------------------------------------
// Workspace layout (bytes):
//   [0,          64 MB)  A bf16              B*N*N*2   = 67,108,864
//   [64 MB,      68 MB)  X0t bf16 col-major  B*CC*N*2  =  4,194,304
//   [68 MB,      72 MB)  X1t bf16 col-major  B*CC*N*2  =  4,194,304
//   [72 MB,      80 MB)  Y   f32  col-major  B*CC*N*4  =  8,388,608
// ---------------------------------------------------------------------------
extern "C" void kernel_launch(void* const* d_in, const int* in_sizes, int n_in,
                              void* d_out, int out_size, void* d_ws, size_t ws_size,
                              hipStream_t stream) {
  const float* X     = (const float*)d_in[0];
  const float* A     = (const float*)d_in[1];
  const float* Theta = (const float*)d_in[2];
  const float* bias  = (const float*)d_in[3];
  float* out = (float*)d_out;

  char* ws = (char*)d_ws;
  __bf16* Abf = (__bf16*)(ws);
  __bf16* X0t = (__bf16*)(ws + (size_t)67108864);
  __bf16* X1t = (__bf16*)(ws + (size_t)71303168);
  float*  Yt  = (float*) (ws + (size_t)75497472);

  // 1) A fp32 -> bf16 (33,554,432 elems / 4 per thread)
  convertA_kernel<<<32768, 256, 0, stream>>>(A, Abf, 8388608);
  // 2) X fp32 -> bf16 column-major (2,097,152 elems)
  convertX_kernel<<<8192, 256, 0, stream>>>(X, X0t);

  // 3) x1 = A @ x0   (bf16 out feeds pass 2 and the combine)
  dim3 g(NN / 16 / 8, CC / COLS_PER_WG, BB);   // 32 x 2 x 2 = 128 WGs
  gemm_diffuse_kernel<<<g, 256, 0, stream>>>(Abf, X0t, X1t, nullptr);
  // 4) y  = A @ x1   (f32 out for the 2y - x0 term)
  gemm_diffuse_kernel<<<g, 256, 0, stream>>>(Abf, X1t, nullptr, Yt);

  // 5) out = relu([x0|x1|2y-x0] @ Theta + bias)
  combine_kernel<<<(BB * TT * NN) / 256, 256, 0, stream>>>(X, X1t, Yt, Theta, bias, out);
}
